// TrafficCCNN_40578851013135
// MI455X (gfx1250) — matrix operand order repaired and checked by reading, not compiled, not run on hardware
//
#include <hip/hip_runtime.h>
#include <hip/hip_bf16.h>

// Model constants (from the reference)
#define N0c   4096
#define N1c   12288
#define WINc  24
#define Hc    64
#define BATCHc 4
#define PREDc 12
#define SLOPEc 0.2f
#define LOG2Ec 1.4426950408889634f

typedef __attribute__((ext_vector_type(16))) _Float16 v16h;
typedef __attribute__((ext_vector_type(8)))  float    v8f;

// ---------------------------------------------------------------------------
// WMMA helpers (CDNA5 v_wmma_f32_16x16x32_f16, wave32)
// ---------------------------------------------------------------------------
__device__ __forceinline__ v8f wmma_f16(v16h a, v16h b, v8f c) {
  // 8 args: (neg_a, A, neg_b, B, c_mod, C, reuse_a, reuse_b)
  return __builtin_amdgcn_wmma_f32_16x16x32_f16(false, a, false, b, (short)0, c,
                                                false, false);
}

// A-fragment (16x32 f16) from row-major f32 matrix X[row][64].
// Lane L: M = L%16; half h<8 -> K = kc*32 + (L/16)*8 + h; h>=8 -> +16.
__device__ __forceinline__ v16h load_a_f32g(const float* __restrict__ X,
                                            int row0, int kc) {
  const int lane = threadIdx.x & 31;
  const float* p = X + (size_t)(row0 + (lane & 15)) * 64 + kc * 32 + ((lane >> 4) << 3);
  v16h a;
#pragma unroll
  for (int i = 0; i < 8; ++i) a[i] = (_Float16)p[i];
#pragma unroll
  for (int i = 0; i < 8; ++i) a[8 + i] = (_Float16)p[16 + i];
  return a;
}

// A-fragment from f16 LDS tile Hs[16][64]
__device__ __forceinline__ v16h load_a_f16s(const _Float16* Hs, int kc) {
  const int lane = threadIdx.x & 31;
  const _Float16* p = Hs + (lane & 15) * 64 + kc * 32 + ((lane >> 4) << 3);
  v16h a;
#pragma unroll
  for (int i = 0; i < 8; ++i) a[i] = p[i];
#pragma unroll
  for (int i = 0; i < 8; ++i) a[8 + i] = p[16 + i];
  return a;
}

// B-fragment (32x16 f16) from LDS weight stored transposed: Wt[n][k], k-dim
// contiguous (ld halves wide).  Lane L: N = nb*16 + L%16; h<8 -> K = kc*32 + (L/16)*8 + h; h>=8 -> +16.
__device__ __forceinline__ v16h load_b_frag(const _Float16* Wt, int nb, int kc, int ld) {
  const int lane = threadIdx.x & 31;
  const _Float16* p = Wt + (size_t)(nb * 16 + (lane & 15)) * ld + kc * 32 + ((lane >> 4) << 3);
  v16h b;
#pragma unroll
  for (int i = 0; i < 8; ++i) b[i] = p[i];
#pragma unroll
  for (int i = 0; i < 8; ++i) b[8 + i] = p[16 + i];
  return b;
}

// ---------------------------------------------------------------------------
// Fast activations straight on the HW TRANS pipe (no libm slow paths).
// Gate magnitudes here are O(1), so range reduction is unnecessary.
// ---------------------------------------------------------------------------
__device__ __forceinline__ float fast_exp2(float x) {
#if __has_builtin(__builtin_amdgcn_exp2f)
  return __builtin_amdgcn_exp2f(x);
#else
  return exp2f(x);
#endif
}
__device__ __forceinline__ float fast_rcp(float x) {
#if __has_builtin(__builtin_amdgcn_rcpf)
  return __builtin_amdgcn_rcpf(x);
#else
  return 1.0f / x;
#endif
}
__device__ __forceinline__ float sigf(float x) {
  return fast_rcp(1.0f + fast_exp2(-x * LOG2Ec));
}
__device__ __forceinline__ float tanh_fast(float x) {
#if __has_builtin(__builtin_amdgcn_tanhf)
  return __builtin_amdgcn_tanhf(x);   // native v_tanh_f32 on gfx1250
#else
  return 1.0f - 2.0f * fast_rcp(1.0f + fast_exp2(2.0f * LOG2Ec * x));
#endif
}
__device__ __forceinline__ float lrelu(float x) { return x > 0.0f ? x : SLOPEc * x; }

// ---------------------------------------------------------------------------
// LSTM: 16 sequences per 256-thread block (8 waves).  W_hh lives in LDS as f16;
// each wave keeps its 4 B-fragments (2 N-tiles x 2 K-chunks, covering 32 of the
// 256 gate columns) in VGPRs across all 24 timesteps.  h round-trips via a 2KB
// f16 LDS tile that feeds the A-fragments.
// ---------------------------------------------------------------------------
__global__ __launch_bounds__(256) void lstm_kernel(
    const float* __restrict__ x, const float* __restrict__ w_ih,
    const float* __restrict__ w_hh, const float* __restrict__ b_ih,
    const float* __restrict__ b_hh, float* __restrict__ h_out) {
  __shared__ _Float16 whh_s[256 * 64];   // [gate j][k]  (already W_hh row-major)
  __shared__ _Float16 hbuf[16 * 64];     // h state, f16, A-source
  __shared__ float gbuf[16 * 256];       // gate pre-activations
  __shared__ float wih_s[256];
  __shared__ float bias_s[256];

  const int tid = threadIdx.x;
  const int wave = tid >> 5;
  const int lane = tid & 31;

  for (int i = tid; i < 256 * 64; i += 256) whh_s[i] = (_Float16)w_hh[i];
  if (tid < 256) { wih_s[tid] = w_ih[tid]; bias_s[tid] = b_ih[tid] + b_hh[tid]; }
  for (int i = tid; i < 16 * 64; i += 256) hbuf[i] = (_Float16)0.0f;
  __syncthreads();

  // Persistent B fragments: wave w handles gate columns [32w, 32w+32)
  v16h bfrag[2][2];
#pragma unroll
  for (int j = 0; j < 2; ++j)
#pragma unroll
    for (int kc = 0; kc < 2; ++kc)
      bfrag[j][kc] = load_b_frag(whh_s, 2 * wave + j, kc, 64);

  float creg[4], hreg[4];
#pragma unroll
  for (int u = 0; u < 4; ++u) { creg[u] = 0.0f; hreg[u] = 0.0f; }

  const int blk = blockIdx.x;
  const int b = blk >> 8;                 // 256 blocks per batch sample
  const int node0 = (blk & 255) << 4;     // 16 sensors per block
  const float* xb = x + (size_t)b * WINc * N0c + node0;

  const int n_d = lane & 15;
  const int mh_d = (lane >> 4) << 3;

  for (int t = 0; t < WINc; ++t) {
    v16h a0 = load_a_f16s(hbuf, 0);
    v16h a1 = load_a_f16s(hbuf, 1);
#pragma unroll
    for (int j = 0; j < 2; ++j) {
      v8f acc = {};
      acc = wmma_f16(a0, bfrag[j][0], acc);
      acc = wmma_f16(a1, bfrag[j][1], acc);
      const int col = wave * 32 + j * 16 + n_d;
#pragma unroll
      for (int r = 0; r < 8; ++r) gbuf[(mh_d + r) * 256 + col] = acc[r];
    }
    __syncthreads();
#pragma unroll
    for (int u = 0; u < 4; ++u) {
      const int idx = tid * 4 + u;
      const int seq = idx >> 6, j = idx & 63;
      const float xt = xb[(size_t)t * N0c + seq];
      const float gi = gbuf[seq * 256 + j]       + xt * wih_s[j]       + bias_s[j];
      const float gf = gbuf[seq * 256 + 64 + j]  + xt * wih_s[64 + j]  + bias_s[64 + j];
      const float gg = gbuf[seq * 256 + 128 + j] + xt * wih_s[128 + j] + bias_s[128 + j];
      const float go = gbuf[seq * 256 + 192 + j] + xt * wih_s[192 + j] + bias_s[192 + j];
      const float c = sigf(gf) * creg[u] + sigf(gi) * tanh_fast(gg);
      creg[u] = c;
      hreg[u] = sigf(go) * tanh_fast(c);
    }
    __syncthreads();
#pragma unroll
    for (int u = 0; u < 4; ++u) hbuf[tid * 4 + u] = (_Float16)hreg[u];
    __syncthreads();
  }
#pragma unroll
  for (int u = 0; u < 4; ++u) {
    const int idx = tid * 4 + u;
    const int seq = idx >> 6, j = idx & 63;
    h_out[((size_t)b * N0c + node0 + seq) * 64 + j] = hreg[u];
  }
}

// ---------------------------------------------------------------------------
// Y[rows,64] = act(X[rows,64] @ W[64,64] + bias).  8 waves x 16 rows = 128
// rows per block; 8 WMMAs per wave.  W transposed into LDS as f16 once.
// ---------------------------------------------------------------------------
__global__ __launch_bounds__(256) void gemm64_kernel(
    const float* __restrict__ X, const float* __restrict__ W,
    const float* __restrict__ bias, float* __restrict__ Y, int rows, int do_relu) {
  __shared__ _Float16 wt_s[64 * 64];  // Wt[n][k]
  __shared__ float bias_s[64];
  const int tid = threadIdx.x;
  for (int i = tid; i < 64 * 64; i += 256) {
    const int n = i >> 6, k = i & 63;
    wt_s[i] = (_Float16)W[k * 64 + n];
  }
  if (tid < 64) bias_s[tid] = bias ? bias[tid] : 0.0f;
  __syncthreads();

  const int wave = tid >> 5;
  const int row0 = blockIdx.x * 128 + wave * 16;
  if (row0 >= rows) return;  // wave-uniform: EXEC stays all-ones for WMMA
  const int lane = tid & 31;
  const v16h a0 = load_a_f32g(X, row0, 0);
  const v16h a1 = load_a_f32g(X, row0, 1);
  const int n_d = lane & 15, mh_d = (lane >> 4) << 3;
#pragma unroll
  for (int nb = 0; nb < 4; ++nb) {
    v8f acc = {};
    acc = wmma_f16(a0, load_b_frag(wt_s, nb, 0, 64), acc);
    acc = wmma_f16(a1, load_b_frag(wt_s, nb, 1, 64), acc);
    const int col = nb * 16 + n_d;
    const float bv = bias_s[col];
#pragma unroll
    for (int r = 0; r < 8; ++r) {
      float v = acc[r] + bv;
      if (do_relu) v = fmaxf(v, 0.0f);
      Y[(size_t)(row0 + mh_d + r) * 64 + col] = v;
    }
  }
}

// ---------------------------------------------------------------------------
// Head: out[b][p][node] = Z[b*N0+node] @ dw2[64,12] + db2  (N padded to 16)
// ---------------------------------------------------------------------------
__global__ __launch_bounds__(256) void head_kernel(
    const float* __restrict__ Z, const float* __restrict__ dw2,
    const float* __restrict__ db2, float* __restrict__ out, int rows) {
  __shared__ _Float16 wt_s[16 * 64];
  __shared__ float b2_s[16];
  const int tid = threadIdx.x;
  for (int i = tid; i < 16 * 64; i += 256) {
    const int n = i >> 6, k = i & 63;
    wt_s[i] = (n < PREDc) ? (_Float16)dw2[k * PREDc + n] : (_Float16)0.0f;
  }
  if (tid < 16) b2_s[tid] = (tid < PREDc) ? db2[tid] : 0.0f;
  __syncthreads();

  const int wave = tid >> 5;
  const int row0 = blockIdx.x * 128 + wave * 16;
  if (row0 >= rows) return;
  const int lane = tid & 31;
  const v16h a0 = load_a_f32g(Z, row0, 0);
  const v16h a1 = load_a_f32g(Z, row0, 1);
  v8f acc = {};
  acc = wmma_f16(a0, load_b_frag(wt_s, 0, 0, 64), acc);
  acc = wmma_f16(a1, load_b_frag(wt_s, 0, 1, 64), acc);
  const int n_d = lane & 15, mh_d = (lane >> 4) << 3;
  if (n_d < PREDc) {
#pragma unroll
    for (int r = 0; r < 8; ++r) {
      const int row = row0 + mh_d + r;      // row = b*N0 + node
      const int b = row >> 12, node = row & (N0c - 1);
      out[((size_t)b * PREDc + n_d) * N0c + node] = acc[r] + b2_s[n_d];
    }
  }
}

// ---------------------------------------------------------------------------
// Attention scalar kernels.  e_nnz = lrelu(prow[row] + pcol[col]); row-norm is
// a guarded divide by per-segment sums (0/0 -> 0, the intended semantics for
// the all-zero branches).
// ---------------------------------------------------------------------------
__global__ void dotpq_kernel(const float* __restrict__ M, const float* __restrict__ avec,
                             float* __restrict__ p, float* __restrict__ q, int rows) {
  const int r = blockIdx.x * blockDim.x + threadIdx.x;
  if (r >= rows) return;
  const float* m = M + (size_t)r * 64;
  float sp = 0.0f, sq = 0.0f;
#pragma unroll
  for (int k = 0; k < 64; ++k) { const float v = m[k]; sp += v * avec[k]; sq += v * avec[64 + k]; }
  if (p) p[r] = sp;
  if (q) q[r] = sq;
}

__global__ void attsum_kernel(const int* __restrict__ rows, const int* __restrict__ cols,
                              const float* __restrict__ prow, const float* __restrict__ pcol,
                              float* __restrict__ sums, int nnz, int nR, int nC, int seg_is_row) {
  const int g = blockIdx.x * blockDim.x + threadIdx.x;
  if (g >= nnz * BATCHc) return;
  const int b = g / nnz, e = g - b * nnz;
  const int r = rows[e], c = cols[e];
  float x = 0.0f;
  if (prow) x += prow[b * nR + r];
  if (pcol) x += pcol[b * nC + c];
  const int seg = seg_is_row ? (b * nR + r) : (b * nC + c);
  atomicAdd(&sums[seg], lrelu(x));
}

__global__ void attscat_kernel(const int* __restrict__ rows, const int* __restrict__ cols,
                               const float* __restrict__ vals,
                               const float* __restrict__ prow, const float* __restrict__ pcol,
                               const float* __restrict__ sums, const float* __restrict__ src,
                               float* __restrict__ out, int nnz, int nR, int nC, int seg_is_row) {
  const int g = blockIdx.x * blockDim.x + threadIdx.x;
  if (g >= nnz * BATCHc * 64) return;
  const int hh = g & 63;
  const int eb = g >> 6;
  const int b = eb / nnz, e = eb - b * nnz;
  const int r = rows[e], c = cols[e];
  int segi, srci;
  if (seg_is_row) { segi = b * nR + r; srci = b * nC + c; }
  else            { segi = b * nC + c; srci = b * nR + r; }
  const float* sp = &src[(size_t)srci * 64 + hh];
  __builtin_prefetch(sp, 0, 0);   // -> global_prefetch_b8 on the gathered row
  float x = 0.0f;
  if (prow) x += prow[b * nR + r];
  if (pcol) x += pcol[b * nC + c];
  const float ev = lrelu(x);
  const float s = sums[segi];
  const float att = (s != 0.0f) ? ev * fast_rcp(s) * vals[e] : 0.0f;
  atomicAdd(&out[(size_t)segi * 64 + hh], att * sp[0]);
}

__global__ void zero_kernel(float* p, int n) {
  for (int i = blockIdx.x * blockDim.x + threadIdx.x; i < n; i += gridDim.x * blockDim.x)
    p[i] = 0.0f;
}
__global__ void relu_kernel(float* p, int n) {
  for (int i = blockIdx.x * blockDim.x + threadIdx.x; i < n; i += gridDim.x * blockDim.x)
    p[i] = fmaxf(p[i], 0.0f);
}
__global__ void reluadd_kernel(const float* __restrict__ a, const float* __restrict__ b,
                               float* __restrict__ o, int n) {
  for (int i = blockIdx.x * blockDim.x + threadIdx.x; i < n; i += gridDim.x * blockDim.x)
    o[i] = fmaxf(a[i], 0.0f) + fmaxf(b[i], 0.0f);
}

// ---------------------------------------------------------------------------
extern "C" void kernel_launch(void* const* d_in, const int* in_sizes, int n_in,
                              void* d_out, int out_size, void* d_ws, size_t ws_size,
                              hipStream_t stream) {
  (void)n_in; (void)out_size; (void)ws_size;
  const float* x    = (const float*)d_in[0];
  const float* w_ih = (const float*)d_in[1];
  const float* w_hh = (const float*)d_in[2];
  const float* b_ih = (const float*)d_in[3];
  const float* b_hh = (const float*)d_in[4];
  const float* dw1  = (const float*)d_in[5];
  const float* db1  = (const float*)d_in[6];
  const float* dw2  = (const float*)d_in[7];
  const float* db2  = (const float*)d_in[8];
  // hmc_params dict flattened in sorted-key order (jax pytree convention)
  const float* hbns01_l1_a  = (const float*)d_in[9];
  const float* hbns01_l1_wt = (const float*)d_in[11];
  const float* hbns01_l2_a  = (const float*)d_in[12];
  const float* hbns01_l2_ws = (const float*)d_in[13];
  const float* hbns01_l2_wt = (const float*)d_in[14];
  const float* hbs0_l1_a = (const float*)d_in[21];
  const float* hbs0_l1_w = (const float*)d_in[22];
  const float* hbs0_l2_a = (const float*)d_in[23];
  const float* hbs0_l2_w = (const float*)d_in[24];
  // topo dict sorted: a0, a1, b1, b2, coa2 -> (rows, cols, vals) each
  const int*   a0r = (const int*)d_in[29];
  const int*   a0c = (const int*)d_in[30];
  const float* a0v = (const float*)d_in[31];
  const int*   b1r = (const int*)d_in[35];
  const int*   b1c = (const int*)d_in[36];
  const float* b1v = (const float*)d_in[37];
  const int nnz_a0 = in_sizes[29];
  const int nnz_b1 = in_sizes[35];

  float* out = (float*)d_out;
  float* ws = (float*)d_ws;

  // workspace layout (floats)
  size_t off = 0;
  const size_t SZ0 = (size_t)BATCHc * N0c * 64;   // 1 M
  const size_t SZ1 = (size_t)BATCHc * N1c * 64;   // 3 M
  float* H0   = ws + off; off += SZ0;   // x0 (LSTM output)
  float* MBUF = ws + off; off += SZ1;   // generic m / sm buffer (max N1 rows)
  float* X00  = ws + off; off += SZ0;   // x0l
  float* X01  = ws + off; off += SZ1;   // x1l
  float* TM   = ws + off; off += SZ0;   // tm buffers
  float* Y00  = ws + off; off += SZ0;
  float* Y10  = ws + off; off += SZ0;
  float* X0F  = ws + off; off += SZ0;
  float* ZB   = ws + off; off += SZ0;
  float* PB   = ws + off; off += (size_t)BATCHc * N1c;
  float* QB   = ws + off; off += (size_t)BATCHc * N1c;
  float* SB   = ws + off; off += (size_t)BATCHc * N1c;

  const int T = 256;
  auto cdiv = [](int a, int b) { return (a + b - 1) / b; };
  const int rows0 = BATCHc * N0c;   // 16384
  const int rows1 = BATCHc * N1c;   // 49152

  // 1) LSTM -> H0  (x0 features)
  lstm_kernel<<<BATCHc * (N0c / 16), T, 0, stream>>>(x, w_ih, w_hh, b_ih, b_hh, H0);

  // 2) x00 = hbs(x0, hbs0_l1) over a0
  gemm64_kernel<<<rows0 / 128, T, 0, stream>>>(H0, hbs0_l1_w, nullptr, MBUF, rows0, 0);
  dotpq_kernel<<<cdiv(rows0, T), T, 0, stream>>>(MBUF, hbs0_l1_a, PB, QB, rows0);
  zero_kernel<<<256, T, 0, stream>>>(SB, BATCHc * N0c);
  zero_kernel<<<2048, T, 0, stream>>>(X00, (int)SZ0);
  attsum_kernel<<<cdiv(nnz_a0 * BATCHc, T), T, 0, stream>>>(a0r, a0c, PB, QB, SB, nnz_a0, N0c, N0c, 1);
  attscat_kernel<<<cdiv(nnz_a0 * BATCHc * 64, T), T, 0, stream>>>(a0r, a0c, a0v, PB, QB, SB, MBUF, X00, nnz_a0, N0c, N0c, 1);
  relu_kernel<<<2048, T, 0, stream>>>(X00, (int)SZ0);   // x0l

  // 3) x01 = on_s of hbns(x1=0, x0) over b1 -> edge features (x1l)
  gemm64_kernel<<<rows0 / 128, T, 0, stream>>>(H0, hbns01_l1_wt, nullptr, TM, rows0, 0);
  dotpq_kernel<<<cdiv(rows0, T), T, 0, stream>>>(TM, hbns01_l1_a, PB, nullptr, rows0);
  zero_kernel<<<256, T, 0, stream>>>(SB, BATCHc * N1c);
  zero_kernel<<<4096, T, 0, stream>>>(X01, (int)SZ1);
  attsum_kernel<<<cdiv(nnz_b1 * BATCHc, T), T, 0, stream>>>(b1r, b1c, PB, nullptr, SB, nnz_b1, N0c, N1c, 0);
  attscat_kernel<<<cdiv(nnz_b1 * BATCHc * 64, T), T, 0, stream>>>(b1r, b1c, b1v, PB, nullptr, SB, TM, X01, nnz_b1, N0c, N1c, 0);
  relu_kernel<<<4096, T, 0, stream>>>(X01, (int)SZ1);   // x1l

  // 4) y00 = hbs(x0l, hbs0_l2) over a0  (relu deferred)
  gemm64_kernel<<<rows0 / 128, T, 0, stream>>>(X00, hbs0_l2_w, nullptr, MBUF, rows0, 0);
  dotpq_kernel<<<cdiv(rows0, T), T, 0, stream>>>(MBUF, hbs0_l2_a, PB, QB, rows0);
  zero_kernel<<<256, T, 0, stream>>>(SB, BATCHc * N0c);
  zero_kernel<<<2048, T, 0, stream>>>(Y00, (int)SZ0);
  attsum_kernel<<<cdiv(nnz_a0 * BATCHc, T), T, 0, stream>>>(a0r, a0c, PB, QB, SB, nnz_a0, N0c, N0c, 1);
  attscat_kernel<<<cdiv(nnz_a0 * BATCHc * 64, T), T, 0, stream>>>(a0r, a0c, a0v, PB, QB, SB, MBUF, Y00, nnz_a0, N0c, N0c, 1);

  // 5) y10 = on_t of hbns(x1l, x0l) over b1 (relu deferred)
  gemm64_kernel<<<rows1 / 128, T, 0, stream>>>(X01, hbns01_l2_ws, nullptr, MBUF, rows1, 0);  // sm (edges)
  gemm64_kernel<<<rows0 / 128, T, 0, stream>>>(X00, hbns01_l2_wt, nullptr, TM, rows0, 0);    // tm (nodes)
  dotpq_kernel<<<cdiv(rows1, T), T, 0, stream>>>(MBUF, hbns01_l2_a, PB, nullptr, rows1);     // ps by edge
  dotpq_kernel<<<cdiv(rows0, T), T, 0, stream>>>(TM, hbns01_l2_a, nullptr, QB, rows0);       // qt by node
  zero_kernel<<<256, T, 0, stream>>>(SB, BATCHc * N0c);
  zero_kernel<<<2048, T, 0, stream>>>(Y10, (int)SZ0);
  // e_st = lrelu(qt[row(node)] + ps[col(edge)]), segment by row; out[node] += att*sm[edge]
  attsum_kernel<<<cdiv(nnz_b1 * BATCHc, T), T, 0, stream>>>(b1r, b1c, QB, PB, SB, nnz_b1, N0c, N1c, 1);
  attscat_kernel<<<cdiv(nnz_b1 * BATCHc * 64, T), T, 0, stream>>>(b1r, b1c, b1v, QB, PB, SB, MBUF, Y10, nnz_b1, N0c, N1c, 1);

  // 6) x0_out = relu(relu(y00)+relu(y10)) = relu(y00)+relu(y10)
  reluadd_kernel<<<2048, T, 0, stream>>>(Y00, Y10, X0F, (int)SZ0);

  // 7) dense head
  gemm64_kernel<<<rows0 / 128, T, 0, stream>>>(X0F, dw1, db1, ZB, rows0, 1);
  head_kernel<<<rows0 / 128, T, 0, stream>>>(ZB, dw2, db2, out, rows0);
}
